// RecurrentGNN_44684839747779
// MI455X (gfx1250) — compile-verified, matrix operand order checked by reading, not run on hardware
//
#include <hip/hip_runtime.h>

typedef _Float16 v16h __attribute__((ext_vector_type(16)));
typedef _Float16 h8  __attribute__((ext_vector_type(8)));
typedef float    v8f __attribute__((ext_vector_type(8)));
typedef float    f8v __attribute__((ext_vector_type(8)));

#define N_NODES  50000
#define N_PAD    50048      // 391 * 128
#define N_EDGES  400000
#define N_GRAPHS 100
#define REPS     10
#define LSTM_H   128

// ---------------------------------------------------------------- utilities
__global__ void k_zero(float* p, int n) {
    int i = blockIdx.x * blockDim.x + threadIdx.x;
    if (i < n) p[i] = 0.f;
}

// ---------------------------------------------------------------- scatter (edge message passing)
__global__ void k_scatter4(const float* __restrict__ x, const int* __restrict__ src,
                           const int* __restrict__ dst, const float* __restrict__ w,
                           float* __restrict__ agg, int E, int Cin) {
    int tid = blockIdx.x * blockDim.x + threadIdx.x;
    int chunks = Cin >> 2;
    int e = tid / chunks;
    if (e >= E) return;
    int c = (tid - e * chunks) << 2;
    int s = src[e], d = dst[e];
    float wt = w[e];
    const float4 v = *(const float4*)(x + (size_t)s * Cin + c);
    float* a = agg + (size_t)d * Cin + c;
    atomicAdd(a + 0, v.x * wt);
    atomicAdd(a + 1, v.y * wt);
    atomicAdd(a + 2, v.z * wt);
    atomicAdd(a + 3, v.w * wt);
}

__global__ void k_scatter3(const float* __restrict__ x, const int* __restrict__ src,
                           const int* __restrict__ dst, const float* __restrict__ w,
                           float* __restrict__ agg, int E) {
    int e = blockIdx.x * blockDim.x + threadIdx.x;
    if (e >= E) return;
    int s = src[e], d = dst[e];
    float wt = w[e];
    atomicAdd(&agg[d * 3 + 0], x[s * 3 + 0] * wt);
    atomicAdd(&agg[d * 3 + 1], x[s * 3 + 1] * wt);
    atomicAdd(&agg[d * 3 + 2], x[s * 3 + 2] * wt);
}

// ---------------------------------------------------------------- layer 0 dense (Cin=3, Cout=64)
__global__ void k_dense0(const float* __restrict__ agg, const float* __restrict__ x,
                         const float* __restrict__ Wrel, const float* __restrict__ Wroot,
                         const float* __restrict__ b, float* __restrict__ out) {
    int tid = blockIdx.x * blockDim.x + threadIdx.x;
    if (tid >= N_NODES * 64) return;
    int n = tid >> 6, c = tid & 63;
    float acc = b[c];
#pragma unroll
    for (int k = 0; k < 3; k++)
        acc += agg[n * 3 + k] * Wrel[k * 64 + c] + x[n * 3 + k] * Wroot[k * 64 + c];
    out[(size_t)n * 64 + c] = fmaxf(acc, 0.f);
}

// ---------------------------------------------------------------- weight pack: WpT[n][k] f16, K = 2*Cin
__global__ void k_pack(const float* __restrict__ Wrel, const float* __restrict__ Wroot,
                       _Float16* __restrict__ WpT, int Cin, int Cout) {
    int tid = blockIdx.x * blockDim.x + threadIdx.x;
    int K = 2 * Cin;
    if (tid >= Cout * K) return;
    int n = tid / K, k = tid - n * K;
    float v = (k < Cin) ? Wrel[k * Cout + n] : Wroot[(k - Cin) * Cout + n];
    WpT[(size_t)n * K + k] = (_Float16)v;
}

// ---------------------------------------------------------------- f32 [agg|x] -> f16 A16[N_PAD][K]
__global__ void k_cvt(const float* __restrict__ agg, const float* __restrict__ xin,
                      _Float16* __restrict__ A16, int Cin) {
    const int K = 2 * Cin;
    int tid = blockIdx.x * blockDim.x + threadIdx.x;
    int cpr = K >> 3;
    int row = tid / cpr;
    if (row >= N_PAD) return;
    int k = (tid - row * cpr) << 3;
    h8 hv = {};
    if (row < N_NODES) {
        const float* sp = (k < Cin) ? agg + (size_t)row * Cin + k
                                    : xin + (size_t)row * Cin + (k - Cin);
        f8v fv = *(const f8v*)sp;
        hv = __builtin_convertvector(fv, h8);
    }
    *(h8*)(A16 + (size_t)row * K + k) = hv;
}

// ---------------------------------------------------------------- WMMA GEMM, branchless double-buffered
// out[N,Cout] = relu( A16[N_PAD,K] @ Wp[K,Cout] + bias )
// block 256 thr = 8 waves; block tile 128x128; wave tile 16x128 (8 wmma accs); K step 32.
__global__ __launch_bounds__(256)
void k_gemm_wmma(const _Float16* __restrict__ A16, const _Float16* __restrict__ WpT,
                 const float* __restrict__ bias, float* __restrict__ out,
                 int Nrows, int K, int Cout) {
    __shared__ _Float16 lA[2][128][40];   // 32 K + 8 pad : row = 80B, 16B-aligned chunks
    __shared__ _Float16 lB[2][128][40];

    const int tid = threadIdx.x;
    const int rowBase = blockIdx.x * 128;
    const int colBase = blockIdx.y * 128;
    const int wave = tid >> 5, lane = tid & 31;
    const int lrow = lane & 15, lgrp = lane >> 4;

    v8f acc[8] = {};

    // staging map: each thread owns 16 f16 (32B) of A and of B per K-step
    const int arow = tid >> 1, ahalf = tid & 1;
    const int bcol = tid >> 1, bq = tid & 1;

    const _Float16* aSrc = A16 + (size_t)(rowBase + arow) * K + ahalf * 16;
    const _Float16* bSrc = WpT + (size_t)(colBase + bcol) * K + bq * 16;
    // generic-pointer low 32 bits == LDS byte address (ISA 10.2 aperture rule)
    unsigned aDst[2] = { (unsigned)(size_t)&lA[0][arow][ahalf * 16],
                         (unsigned)(size_t)&lA[1][arow][ahalf * 16] };
    unsigned bDst[2] = { (unsigned)(size_t)&lB[0][bcol][bq * 16],
                         (unsigned)(size_t)&lB[1][bcol][bq * 16] };

    auto stage = [&](int k0, int buf) {
        unsigned long long ga = (unsigned long long)(size_t)(aSrc + k0);
        unsigned long long gb = (unsigned long long)(size_t)(bSrc + k0);
        asm volatile("global_load_async_to_lds_b128 %0, %1, off"
                     :: "v"(aDst[buf]), "v"(ga) : "memory");
        asm volatile("global_load_async_to_lds_b128 %0, %1, off offset:16"
                     :: "v"(aDst[buf]), "v"(ga) : "memory");
        asm volatile("global_load_async_to_lds_b128 %0, %1, off"
                     :: "v"(bDst[buf]), "v"(gb) : "memory");
        asm volatile("global_load_async_to_lds_b128 %0, %1, off offset:16"
                     :: "v"(bDst[buf]), "v"(gb) : "memory");
    };

    stage(0, 0);   // prologue

    for (int k0 = 0; k0 < K; k0 += 32) {
        const int buf = (k0 >> 5) & 1;
        // branchless pipeline: always prefetch next step (over-read <=64B into
        // adjacent owned workspace; lands in the dead LDS buffer; tail copies
        // retired by the implicit wait-idle at ENDPGM).
        stage(k0 + 32, buf ^ 1);
        // ASYNCcnt retires in order: <=4 means this buffer's 4 copies are done,
        // next buffer's 4 still in flight.
        asm volatile("s_wait_asynccnt 0x4" ::: "memory");
        __syncthreads();   // buffer `buf` staged by all waves

        // A fragment: lane holds row wave*16+lrow; elems 0-7 <- K lgrp*8+j, 8-15 <- 16+lgrp*8+j
        const int r = wave * 16 + lrow;
        h8 alo = *(const h8*)&lA[buf][r][lgrp * 8];
        h8 ahi = *(const h8*)&lA[buf][r][16 + lgrp * 8];
        v16h afrag = __builtin_shufflevector(alo, ahi,
            0, 1, 2, 3, 4, 5, 6, 7, 8, 9, 10, 11, 12, 13, 14, 15);

        // preload ALL 8 B fragments...
        v16h bfr[8];
#pragma unroll
        for (int nt = 0; nt < 8; nt++) {
            const int cc = nt * 16 + lrow;
            h8 blo = *(const h8*)&lB[buf][cc][lgrp * 8];
            h8 bhi = *(const h8*)&lB[buf][cc][16 + lgrp * 8];
            bfr[nt] = __builtin_shufflevector(blo, bhi,
                0, 1, 2, 3, 4, 5, 6, 7, 8, 9, 10, 11, 12, 13, 14, 15);
        }
        // ...and force them live so the scheduler issues all ds_loads before the
        // WMMA chain (single DScnt drain instead of 8).
        asm volatile("" : "+v"(bfr[0]), "+v"(bfr[1]), "+v"(bfr[2]), "+v"(bfr[3]),
                          "+v"(bfr[4]), "+v"(bfr[5]), "+v"(bfr[6]), "+v"(bfr[7]),
                          "+v"(afrag));
#pragma unroll
        for (int nt = 0; nt < 8; nt++) {
            acc[nt] = __builtin_amdgcn_wmma_f32_16x16x32_f16(
                false, afrag, false, bfr[nt], (short)0, acc[nt], false, false);
        }
        __syncthreads();   // all reads of `buf` done before it is re-staged
    }

    // D layout: row = rr + 8*lgrp, col = lane&15
#pragma unroll
    for (int nt = 0; nt < 8; nt++) {
        int gcol = colBase + nt * 16 + lrow;
        float bv = bias[gcol];
#pragma unroll
        for (int rr = 0; rr < 8; rr++) {
            int grow = rowBase + wave * 16 + lgrp * 8 + rr;
            if (grow < Nrows) {
                float v = acc[nt][rr] + bv;
                out[(size_t)grow * Cout + gcol] = fmaxf(v, 0.f);
            }
        }
    }
}

// ---------------------------------------------------------------- global mean pool
__global__ void k_pool_add(const float* __restrict__ feat, const int* __restrict__ batch,
                           float* __restrict__ sums) {
    int tid = blockIdx.x * blockDim.x + threadIdx.x;
    if (tid >= N_NODES * 128) return;
    int n = tid >> 7, c = tid & 127;
    atomicAdd(&sums[batch[n] * 128 + c], feat[(size_t)n * 128 + c]);
}
__global__ void k_pool_cnt(const int* __restrict__ batch, float* __restrict__ cnts) {
    int n = blockIdx.x * blockDim.x + threadIdx.x;
    if (n < N_NODES) atomicAdd(&cnts[batch[n]], 1.0f);
}
__global__ void k_pool_div(float* __restrict__ sums, const float* __restrict__ cnts) {
    int tid = blockIdx.x * blockDim.x + threadIdx.x;
    if (tid >= N_GRAPHS * 128) return;
    int g = tid >> 7;
    sums[tid] = sums[tid] / fmaxf(cnts[g], 1.0f);
}

// ---------------------------------------------------------------- LSTM: 10 seqs x 10 steps x 128
__device__ __forceinline__ float sigf(float x) { return 1.0f / (1.0f + __expf(-x)); }

__global__ __launch_bounds__(512)
void k_lstm(const float* __restrict__ pooled, const float* __restrict__ Wih,
            const float* __restrict__ Whh, const float* __restrict__ bih,
            const float* __restrict__ bhh, float* __restrict__ hseq) {
    __shared__ float sh[10][LSTM_H];
    __shared__ float sc[10][LSTM_H];
    __shared__ float sg[10][4 * LSTM_H];
    int tid = threadIdx.x;

    for (int idx = tid; idx < 10 * LSTM_H; idx += 512) {
        sh[idx >> 7][idx & 127] = 0.f;
        sc[idx >> 7][idx & 127] = 0.f;
    }
    __syncthreads();

    for (int t = 0; t < REPS; t++) {
        int j = tid;
        float bj = bih[j] + bhh[j];
        for (int b = 0; b < 10; b++) {
            const float* xr = pooled + (size_t)(b * REPS + t) * LSTM_H;
            float acc = bj;
#pragma unroll 4
            for (int k = 0; k < LSTM_H; k++) acc += Wih[j * LSTM_H + k] * xr[k];
#pragma unroll 4
            for (int k = 0; k < LSTM_H; k++) acc += Whh[j * LSTM_H + k] * sh[b][k];
            sg[b][j] = acc;
        }
        __syncthreads();
        for (int idx = tid; idx < 10 * LSTM_H; idx += 512) {
            int b = idx >> 7, k = idx & 127;
            float ig = sigf(sg[b][k]);
            float fg = sigf(sg[b][LSTM_H + k]);
            float gg = tanhf(sg[b][2 * LSTM_H + k]);
            float og = sigf(sg[b][3 * LSTM_H + k]);
            float cc = fg * sc[b][k] + ig * gg;
            float hh = og * tanhf(cc);
            sc[b][k] = cc;
            sh[b][k] = hh;
            hseq[(size_t)(b * REPS + t) * LSTM_H + k] = hh;
        }
        __syncthreads();
    }
}

// ---------------------------------------------------------------- MLP head (linear; ref discards relu)
__global__ __launch_bounds__(128)
void k_head(const float* __restrict__ hseq, const float* __restrict__ W0, const float* __restrict__ b0,
            const float* __restrict__ W1, const float* __restrict__ b1,
            const float* __restrict__ Wo, const float* __restrict__ bo, float* __restrict__ out) {
    __shared__ float hv[128], m0[128], m1[64];
    int g = blockIdx.x, t = threadIdx.x;
    hv[t] = hseq[(size_t)g * 128 + t];
    __syncthreads();
    float a0 = b0[t];
    for (int k = 0; k < 128; k++) a0 += hv[k] * W0[k * 128 + t];
    m0[t] = a0;
    __syncthreads();
    if (t < 64) {
        float a1 = b1[t];
        for (int k = 0; k < 128; k++) a1 += m0[k] * W1[k * 64 + t];
        m1[t] = a1;
    }
    __syncthreads();
    if (t == 0) {
        float a = bo[0];
        for (int k = 0; k < 64; k++) a += m1[k] * Wo[k];
        out[g] = a;
    }
}

// ---------------------------------------------------------------- launch
extern "C" void kernel_launch(void* const* d_in, const int* in_sizes, int n_in,
                              void* d_out, int out_size, void* d_ws, size_t ws_size,
                              hipStream_t stream) {
    (void)in_sizes; (void)n_in; (void)out_size; (void)ws_size;
    const float* x   = (const float*)d_in[0];
    const int*   ei  = (const int*)d_in[1];
    const int*   src = ei;
    const int*   dst = ei + N_EDGES;
    const float* ew  = (const float*)d_in[2];
    const int*   bat = (const int*)d_in[3];
    // gcn layer l: W_rel=d_in[4+3l], W_root=d_in[5+3l], b_rel=d_in[6+3l]
    const float* Whh = (const float*)d_in[22];
    const float* Wih = (const float*)d_in[23];
    const float* bhh = (const float*)d_in[24];
    const float* bih = (const float*)d_in[25];
    const float* W0  = (const float*)d_in[26];
    const float* b0  = (const float*)d_in[27];
    const float* W1  = (const float*)d_in[28];
    const float* b1  = (const float*)d_in[29];
    const float* Wo  = (const float*)d_in[30];
    const float* bo  = (const float*)d_in[31];

    // workspace layout
    float* featA = (float*)d_ws;                                 // N*256 f32
    float* featB = featA + (size_t)N_NODES * 256;                // N*512 f32
    float* agg   = featB + (size_t)N_NODES * 512;                // N*512 f32
    _Float16* A16 = (_Float16*)(agg + (size_t)N_NODES * 512);    // N_PAD*1024 f16
    _Float16* WpT = A16 + (size_t)N_PAD * 1024;                  // 512*1024 f16
    float* sums  = (float*)(WpT + 512 * 1024);                   // 100*128
    float* cnts  = sums + N_GRAPHS * 128;                        // 100
    float* hseq  = cnts + N_GRAPHS;                              // 100*128

    const int B = 256;
    auto blocks = [](long long n, int b) { return (unsigned)((n + b - 1) / b); };

    // ---- layer 0: Cin=3 -> 64 (scalar) ----
    k_zero<<<blocks(N_NODES * 3, B), B, 0, stream>>>(agg, N_NODES * 3);
    k_scatter3<<<blocks(N_EDGES, B), B, 0, stream>>>(x, src, dst, ew, agg, N_EDGES);
    k_dense0<<<blocks((long long)N_NODES * 64, B), B, 0, stream>>>(
        agg, x, (const float*)d_in[4], (const float*)d_in[5], (const float*)d_in[6], featA);

    // ---- layers 1..5: WMMA path ----
    const int ins[5]  = {64, 128, 256, 512, 256};
    const int outs[5] = {128, 256, 512, 256, 128};
    float* cur = featA;
    float* nxt = featB;
    for (int l = 0; l < 5; l++) {
        int Cin = ins[l], Cout = outs[l], K = 2 * Cin;
        const float* Wrel  = (const float*)d_in[4 + 3 * (l + 1)];
        const float* Wroot = (const float*)d_in[5 + 3 * (l + 1)];
        const float* brel  = (const float*)d_in[6 + 3 * (l + 1)];

        k_zero<<<blocks((long long)N_NODES * Cin, B), B, 0, stream>>>(agg, N_NODES * Cin);
        k_scatter4<<<blocks((long long)N_EDGES * (Cin >> 2), B), B, 0, stream>>>(
            cur, src, dst, ew, agg, N_EDGES, Cin);
        k_pack<<<blocks((long long)Cout * K, B), B, 0, stream>>>(Wrel, Wroot, WpT, Cin, Cout);
        k_cvt<<<blocks((long long)N_PAD * (K >> 3), B), B, 0, stream>>>(agg, cur, A16, Cin);

        dim3 grid(N_PAD / 128, Cout / 128);
        k_gemm_wmma<<<grid, 256, 0, stream>>>(A16, WpT, brel, nxt, N_NODES, K, Cout);
        float* t = cur; cur = nxt; nxt = t;
    }
    // final features in `cur` (featB), 128 channels

    // ---- global mean pool ----
    k_zero<<<blocks(N_GRAPHS * 128 + N_GRAPHS, B), B, 0, stream>>>(sums, N_GRAPHS * 128 + N_GRAPHS);
    k_pool_add<<<blocks((long long)N_NODES * 128, B), B, 0, stream>>>(cur, bat, sums);
    k_pool_cnt<<<blocks(N_NODES, B), B, 0, stream>>>(bat, cnts);
    k_pool_div<<<blocks(N_GRAPHS * 128, B), B, 0, stream>>>(sums, cnts);

    // ---- LSTM ----
    k_lstm<<<1, 512, 0, stream>>>(sums, Wih, Whh, bih, bhh, hseq);

    // ---- head ----
    k_head<<<N_GRAPHS, 128, 0, stream>>>(hseq, W0, b0, W1, b1, Wo, bo, (float*)d_out);
}
